// ComplexGCN_14328010899638
// MI455X (gfx1250) — matrix operand order repaired and checked by reading, not compiled, not run on hardware
//
#include <hip/hip_runtime.h>
#include <hip/hip_bf16.h>

// ComplexGCN pipeline for MI455X (gfx1250, wave32):
//   GCN1 -> relu(MHA) -> GCN2 -> relu(MHA) -> GCN3 -> sigmoid
// All dense matmuls + attention use v_wmma_f32_16x16x32_bf16 (f32 accum).
// WMMA operands are pre-converted packed bf16; attention is flash-style with
// log2-domain online softmax, step-major batched shuffle max-reduction, and
// row-sums computed by an extra WMMA against a constant ones fragment.

#define D_IN   128
#define D_HID  256
#define D_OUT  64
#define NHEAD  4
#define DH     64
#define QKV_D  768

typedef __attribute__((ext_vector_type(16))) __bf16 v16bf;
typedef __attribute__((ext_vector_type(8)))  float  v8f;

union Frag16 {
  unsigned short us[16];
  unsigned int   u32[8];
  v16bf          v;
};

__device__ __forceinline__ unsigned short f2bf(float f) {
  unsigned int u = __float_as_uint(f);
  return (unsigned short)((u + 0x7FFFu + ((u >> 16) & 1u)) >> 16); // RNE
}

// ---------------- f32 -> packed bf16 conversion pass ----------------
__global__ void cvt_bf16_kernel(const float* __restrict__ in,
                                unsigned short* __restrict__ out, int total) {
  int i = blockIdx.x * blockDim.x + threadIdx.x;
  if (i < total) out[i] = f2bf(in[i]);
}

// ---------------- degree / symmetric normalization ----------------
__global__ void deg_init_kernel(float* deg, int n) {
  int i = blockIdx.x * blockDim.x + threadIdx.x;
  if (i < n) deg[i] = 1.0f; // self-loop weight
}
__global__ void deg_accum_kernel(float* deg, const int* __restrict__ col,
                                 const float* __restrict__ ew, int e) {
  int i = blockIdx.x * blockDim.x + threadIdx.x;
  if (i < e) atomicAdd(&deg[col[i]], ew[i]);
}
__global__ void deg_fin_kernel(float* deg, int n) {
  int i = blockIdx.x * blockDim.x + threadIdx.x;
  if (i < n) { float d = deg[i]; deg[i] = (d > 0.f) ? rsqrtf(d) : 0.f; }
}

// ---------------- WMMA GEMM: C[N,M] = act(A[N,K] @ W[M,K]^T + bias) ----------
// A, W are packed bf16 (row-major, K contiguous). One wave per 16x16 tile.
// ACT: 0 = none, 1 = relu.  OUTBF: 0 = f32 out, 1 = bf16 out.
template <int ACT, int OUTBF>
__global__ void gemm_wmma_kernel(const unsigned short* __restrict__ A,
                                 const unsigned short* __restrict__ W,
                                 const float* __restrict__ bias,
                                 void* __restrict__ Cout, int K, int M) {
  const int lane = threadIdx.x;
  const int half = lane >> 4;
  const int lrow = lane & 15;
  const int tm = blockIdx.x * 16; // output col tile
  const int tn = blockIdx.y * 16; // output row tile

  v8f acc = {};
  const unsigned int* arow32 =
      (const unsigned int*)(A + (size_t)(tn + lrow) * K);
  const unsigned int* wrow32 =
      (const unsigned int*)(W + (size_t)(tm + lrow) * K + (half << 4));

  for (int k0 = 0; k0 < K; k0 += 32) {
    Frag16 a, b;
    const int kd = k0 >> 1; // dword index
#pragma unroll
    for (int i = 0; i < 8; ++i) {
      // A-frag dword i covers k pair at 2*(i + 8*half)
      a.u32[i] = arow32[kd + i + (half << 3)];
    }
#pragma unroll
    for (int j = 0; j < 8; ++j) {
      b.u32[j] = wrow32[kd + j]; // contiguous: 2x b128
    }
    acc = __builtin_amdgcn_wmma_f32_16x16x32_bf16(false, a.v, false, b.v,
                                                  (short)0, acc, false, false);
  }

  const int colI = tm + lrow;
  const float bv = bias ? bias[colI] : 0.f;
#pragma unroll
  for (int r = 0; r < 8; ++r) {
    float v = acc[r] + bv;
    if (ACT == 1) v = fmaxf(v, 0.f);
    const size_t idx = (size_t)(tn + r + 8 * half) * M + colI;
    if (OUTBF)
      ((unsigned short*)Cout)[idx] = f2bf(v);
    else
      ((float*)Cout)[idx] = v;
  }
}

// ---------------- flash attention (one wave per 16-query tile per head) ----------
// qkv: packed bf16 [N, 768] = [Q | K | V]; out: packed bf16 [N, 256].
__global__ void attn_flash_kernel(const unsigned short* __restrict__ qkv,
                                  unsigned short* __restrict__ out, int N) {
  const int lane = threadIdx.x;
  const int half = lane >> 4;
  const int lrow = lane & 15;
  const int qbase = blockIdx.x * 16;
  const int h = blockIdx.y;
  const int hq = h * DH;
  const int hk = D_HID + h * DH;
  const int hv = 2 * D_HID + h * DH;

  // log2-domain scale: (1/sqrt(64)) * log2(e)
  const float SC = 0.1803368801111137f;

  __shared__ unsigned short pbuf[16 * 32]; // P tile, bf16, row-major [16][32]

  // Q A-fragments for the two 32-wide K-dim chunks (raw bf16; scale folded into SC).
  Frag16 qa[2];
  {
    const unsigned int* qrow32 =
        (const unsigned int*)(qkv + (size_t)(qbase + lrow) * QKV_D + hq);
#pragma unroll
    for (int kc = 0; kc < 2; ++kc) {
#pragma unroll
      for (int i = 0; i < 8; ++i) {
        qa[kc].u32[i] =
            qrow32[kc * 16 + i + (half << 3)];
      }
    }
  }

  // Constant all-1.0 bf16 B-fragment: rowsum(P) = P @ ones via WMMA.
  Frag16 ones;
#pragma unroll
  for (int i = 0; i < 8; ++i) ones.u32[i] = 0x3F803F80u;

  v8f acc[4] = {v8f{}, v8f{}, v8f{}, v8f{}};
  float m[8], ssum[8];
#pragma unroll
  for (int r = 0; r < 8; ++r) { m[r] = -1e30f; ssum[r] = 0.f; }

  for (int jb = 0; jb < N; jb += 32) {
    // S = Q @ K^T for 16 queries x 32 keys (two 16x16 tiles)
    v8f s01[2];
#pragma unroll
    for (int ct = 0; ct < 2; ++ct) {
      v8f s = {};
#pragma unroll
      for (int kc = 0; kc < 2; ++kc) {
        Frag16 bk;
        const unsigned int* krow32 = (const unsigned int*)(
            qkv + (size_t)(jb + ct * 16 + lrow) * QKV_D + hk + kc * 32 +
            (half << 4));
#pragma unroll
        for (int j = 0; j < 8; ++j) bk.u32[j] = krow32[j];
        s = __builtin_amdgcn_wmma_f32_16x16x32_bf16(false, qa[kc].v, false, bk.v,
                                                    (short)0, s, false, false);
      }
      s01[ct] = s;
    }

    // Step-major batched butterfly row-max across the 16 lanes of each half.
    float mx[8];
#pragma unroll
    for (int r = 0; r < 8; ++r) mx[r] = fmaxf(s01[0][r], s01[1][r]);
#pragma unroll
    for (int st = 1; st < 16; st <<= 1) {
      float o[8];
#pragma unroll
      for (int r = 0; r < 8; ++r) o[r] = __shfl_xor(mx[r], st, 32);
#pragma unroll
      for (int r = 0; r < 8; ++r) mx[r] = fmaxf(mx[r], o[r]);
    }

    // Online softmax in log2 domain; half owns rows {8*half..8*half+7}.
    float alpha[8];
#pragma unroll
    for (int r = 0; r < 8; ++r) {
      float mn = fmaxf(m[r], mx[r] * SC);
      alpha[r] = exp2f(m[r] - mn);
      m[r] = mn;
      float p0 = exp2f(fmaf(s01[0][r], SC, -mn));
      float p1 = exp2f(fmaf(s01[1][r], SC, -mn));
      pbuf[(r + 8 * half) * 32 + lrow]      = f2bf(p0);
      pbuf[(r + 8 * half) * 32 + 16 + lrow] = f2bf(p1);
    }
    __syncthreads();

    // Transpose P (C-layout) into an A-fragment via LDS.
    Frag16 pa;
#pragma unroll
    for (int i = 0; i < 8; ++i) {
      int ko = 2 * i + (half << 4);
      pa.us[2 * i]     = pbuf[lrow * 32 + ko];
      pa.us[2 * i + 1] = pbuf[lrow * 32 + ko + 1];
    }
    __syncthreads();

    // Row-sums of P via WMMA against the ones fragment (replaces shuffle-sum).
    v8f rs = {};
    rs = __builtin_amdgcn_wmma_f32_16x16x32_bf16(false, pa.v, false, ones.v,
                                                 (short)0, rs, false, false);

    // Rescale accumulators + running sums, then acc += P @ V (4 d-chunks of 16).
#pragma unroll
    for (int r = 0; r < 8; ++r) ssum[r] = ssum[r] * alpha[r] + rs[r];
#pragma unroll
    for (int t = 0; t < 4; ++t)
#pragma unroll
      for (int r = 0; r < 8; ++r) acc[t][r] *= alpha[r];

#pragma unroll
    for (int t = 0; t < 4; ++t) {
      Frag16 bv;
      const unsigned short* vptr =
          qkv + (size_t)(jb + (half << 4)) * QKV_D + hv + t * 16 + lrow;
#pragma unroll
      for (int j = 0; j < 8; ++j) {
        bv.us[2 * j]     = vptr[(size_t)(2 * j) * QKV_D];
        bv.us[2 * j + 1] = vptr[(size_t)(2 * j + 1) * QKV_D];
      }
      acc[t] = __builtin_amdgcn_wmma_f32_16x16x32_bf16(false, pa.v, false, bv.v,
                                                       (short)0, acc[t], false, false);
    }
  }

  // Normalize and write O (bf16) into [N, 256] at this head's column block.
#pragma unroll
  for (int r = 0; r < 8; ++r) {
    float inv = (ssum[r] > 0.f) ? 1.f / ssum[r] : 0.f;
    int orow = qbase + r + 8 * half;
#pragma unroll
    for (int t = 0; t < 4; ++t)
      out[(size_t)orow * D_HID + hq + t * 16 + lrow] = f2bf(acc[t][r] * inv);
  }
}

// ---------------- GCN aggregation (f32) ----------------
__global__ void gcn_self_kernel(const float* __restrict__ xw,
                                const float* __restrict__ dinv,
                                const float* __restrict__ bias,
                                float* __restrict__ out, int lf, int total) {
  int gid = blockIdx.x * blockDim.x + threadIdx.x;
  if (gid >= total) return;
  int n = gid >> lf;
  int f = gid & ((1 << lf) - 1);
  float di = dinv[n];
  out[gid] = di * di * xw[gid] + bias[f];
}

__global__ void gcn_scatter_kernel(const float* __restrict__ xw,
                                   const float* __restrict__ dinv,
                                   const int* __restrict__ row,
                                   const int* __restrict__ col,
                                   const float* __restrict__ ew,
                                   float* __restrict__ out, int lf, int total) {
  int gid = blockIdx.x * blockDim.x + threadIdx.x;
  if (gid >= total) return;
  int e = gid >> lf;
  int f = gid & ((1 << lf) - 1);
  int r = row[e], c = col[e];
  float coeff = dinv[r] * ew[e] * dinv[c];
  atomicAdd(&out[((size_t)c << lf) + f], coeff * xw[((size_t)r << lf) + f]);
}

__global__ void sigmoid_kernel(const float* __restrict__ in, float* __restrict__ out,
                               int total) {
  int i = blockIdx.x * blockDim.x + threadIdx.x;
  if (i < total) out[i] = 1.f / (1.f + __expf(-in[i]));
}

// ---------------- host launch ----------------
extern "C" void kernel_launch(void* const* d_in, const int* in_sizes, int n_in,
                              void* d_out, int out_size, void* d_ws, size_t ws_size,
                              hipStream_t stream) {
  const float* x0   = (const float*)d_in[0];
  const float* ew   = (const float*)d_in[1];
  const float* W1   = (const float*)d_in[2];
  const float* b1   = (const float*)d_in[3];
  const float* W2   = (const float*)d_in[4];
  const float* b2   = (const float*)d_in[5];
  const float* W3   = (const float*)d_in[6];
  const float* b3   = (const float*)d_in[7];
  const float* Wqkv = (const float*)d_in[8];
  const float* bqkv = (const float*)d_in[9];
  const float* Wo   = (const float*)d_in[10];
  const float* bo   = (const float*)d_in[11];
  const int*   row  = (const int*)d_in[12];
  const int*   col  = (const int*)d_in[13];

  const int N = in_sizes[0] / D_IN;
  const int E = in_sizes[1];

  // ---- workspace layout ----
  float* dinv = (float*)d_ws;                  // [N]
  float* xwf  = dinv + N;                      // [N,256] f32 gcn x@W^T
  float* aggf = xwf + (size_t)N * D_HID;       // [N,256] f32 gcn aggregate

  unsigned short* w1b   = (unsigned short*)(aggf + (size_t)N * D_HID);
  unsigned short* w2b   = w1b + D_HID * D_IN;        // 256*128
  unsigned short* w3b   = w2b + D_HID * D_HID;       // 256*256
  unsigned short* wqkvb = w3b + D_OUT * D_HID;       // 64*256
  unsigned short* wob   = wqkvb + QKV_D * D_HID;     // 768*256
  unsigned short* x0b   = wob + D_HID * D_HID;       // [N,128]
  unsigned short* xb    = x0b + (size_t)N * D_IN;    // [N,256] gcn out (bf16)
  unsigned short* qkvb  = xb + (size_t)N * D_HID;    // [N,768]
  unsigned short* yb    = qkvb + (size_t)N * QKV_D;  // [N,256] attn out
  unsigned short* zb    = yb + (size_t)N * D_HID;    // [N,256] mha out (relu)

  const dim3 wave(32, 1, 1);
  auto cvt = [&](const float* src, unsigned short* dst, int total) {
    cvt_bf16_kernel<<<(total + 255) / 256, 256, 0, stream>>>(src, dst, total);
  };

  // --- normalization: deg = 1 + sum_in(ew); dinv = rsqrt(deg) ---
  deg_init_kernel<<<(N + 255) / 256, 256, 0, stream>>>(dinv, N);
  deg_accum_kernel<<<(E + 255) / 256, 256, 0, stream>>>(dinv, col, ew, E);
  deg_fin_kernel<<<(N + 255) / 256, 256, 0, stream>>>(dinv, N);

  // --- one-time bf16 conversions ---
  cvt(W1, w1b, D_HID * D_IN);
  cvt(W2, w2b, D_HID * D_HID);
  cvt(W3, w3b, D_OUT * D_HID);
  cvt(Wqkv, wqkvb, QKV_D * D_HID);
  cvt(Wo, wob, D_HID * D_HID);
  cvt(x0, x0b, N * D_IN);

  // --- GCN1: x0b[N,128] @ W1^T -> xwf; aggregate -> aggf; -> xb (bf16) ---
  gemm_wmma_kernel<0, 0><<<dim3(D_HID / 16, N / 16), wave, 0, stream>>>(x0b, w1b, nullptr, xwf, D_IN, D_HID);
  gcn_self_kernel<<<(N * D_HID) / 256, 256, 0, stream>>>(xwf, dinv, b1, aggf, 8, N * D_HID);
  gcn_scatter_kernel<<<(E << 8) / 256, 256, 0, stream>>>(xwf, dinv, row, col, ew, aggf, 8, E << 8);
  cvt(aggf, xb, N * D_HID);

  // --- MHA1: xb -> qkvb -> attn(yb) -> relu(proj) -> zb ---
  gemm_wmma_kernel<0, 1><<<dim3(QKV_D / 16, N / 16), wave, 0, stream>>>(xb, wqkvb, bqkv, qkvb, D_HID, QKV_D);
  attn_flash_kernel<<<dim3(N / 16, NHEAD), wave, 0, stream>>>(qkvb, yb, N);
  gemm_wmma_kernel<1, 1><<<dim3(D_HID / 16, N / 16), wave, 0, stream>>>(yb, wob, bo, zb, D_HID, D_HID);

  // --- GCN2: zb @ W2^T -> xwf; aggregate -> aggf; -> xb ---
  gemm_wmma_kernel<0, 0><<<dim3(D_HID / 16, N / 16), wave, 0, stream>>>(zb, w2b, nullptr, xwf, D_HID, D_HID);
  gcn_self_kernel<<<(N * D_HID) / 256, 256, 0, stream>>>(xwf, dinv, b2, aggf, 8, N * D_HID);
  gcn_scatter_kernel<<<(E << 8) / 256, 256, 0, stream>>>(xwf, dinv, row, col, ew, aggf, 8, E << 8);
  cvt(aggf, xb, N * D_HID);

  // --- MHA2 ---
  gemm_wmma_kernel<0, 1><<<dim3(QKV_D / 16, N / 16), wave, 0, stream>>>(xb, wqkvb, bqkv, qkvb, D_HID, QKV_D);
  attn_flash_kernel<<<dim3(N / 16, NHEAD), wave, 0, stream>>>(qkvb, yb, N);
  gemm_wmma_kernel<1, 1><<<dim3(D_HID / 16, N / 16), wave, 0, stream>>>(yb, wob, bo, zb, D_HID, D_HID);

  // --- GCN3: zb @ W3^T -> xwf[N,64]; aggregate -> aggf; sigmoid -> d_out ---
  gemm_wmma_kernel<0, 0><<<dim3(D_OUT / 16, N / 16), wave, 0, stream>>>(zb, w3b, nullptr, xwf, D_HID, D_OUT);
  gcn_self_kernel<<<(N * D_OUT) / 256, 256, 0, stream>>>(xwf, dinv, b3, aggf, 6, N * D_OUT);
  gcn_scatter_kernel<<<(E << 6) / 256, 256, 0, stream>>>(xwf, dinv, row, col, ew, aggf, 6, E << 6);
  sigmoid_kernel<<<(N * D_OUT) / 256, 256, 0, stream>>>(aggf, (float*)d_out, N * D_OUT);
}
